// TableGNN_55843164782679
// MI455X (gfx1250) — matrix operand order repaired and checked by reading, not compiled, not run on hardware
//
#include <hip/hip_runtime.h>
#include <hip/hip_bf16.h>

// ---------------- WMMA vector types (gfx1250, wave32) ----------------
typedef __attribute__((ext_vector_type(16))) _Float16 v16h;
typedef __attribute__((ext_vector_type(8)))  _Float16 v8h;
typedef __attribute__((ext_vector_type(8)))  float    v8f;

#define NEG_SLOPE 0.2f
#define LDS_STRIDE 136   // 128 + 8 halves pad: 272B = 68 dwords -> bank stride 4

// ---------------------------------------------------------------------
// Float atomic max via integer atomics (native global_atomic_max/min).
// Works for mixed-sign IEEE floats with init = -inf.
__device__ __forceinline__ void atomicMaxF(float* addr, float val) {
    if (val >= 0.0f) {
        atomicMax((int*)addr, __float_as_int(val));
    } else {
        atomicMin((unsigned int*)addr, __float_as_uint(val));
    }
}

// ---------------------------------------------------------------------
// Weight transpose + f32->f16 convert:  WT[(n+colOff)*128 + k] = W[k*ncols + n]
// WT is column-major (contiguous K per column) so B-fragments are two
// contiguous 16B loads per lane.
__global__ void transposeConvertK(const float* __restrict__ W,
                                  _Float16* __restrict__ WT,
                                  int ncols, int colOff) {
    int t = blockIdx.x * blockDim.x + threadIdx.x;
    if (t >= ncols * 128) return;
    int n = t >> 7;        // column of W
    int k = t & 127;       // row of W
    WT[(size_t)(n + colOff) * 128 + k] = (_Float16)W[(size_t)k * ncols + n];
}

__global__ void concatBiasK(const float* __restrict__ b0,
                            const float* __restrict__ b1,
                            float* __restrict__ dst) {
    int t = threadIdx.x;
    if (t < 128) dst[t] = (t < 64) ? b0[t] : b1[t - 64];
}

// ---------------------------------------------------------------------
// Encoder stage 1:  tmp16[N,128] = (f16) relu(x[N,8] @ ce_w1[8,128] + b1)
__global__ void enc1K(const float* __restrict__ x,
                      const float* __restrict__ w1,
                      const float* __restrict__ b1,
                      _Float16* __restrict__ tmp16, int N) {
    int t = blockIdx.x * blockDim.x + threadIdx.x;
    if (t >= N * 128) return;
    int i = t >> 7, c = t & 127;
    const float* xi = x + (size_t)i * 8;
    float acc = b1[c];
#pragma unroll
    for (int k = 0; k < 8; ++k) acc += xi[k] * w1[k * 128 + c];
    tmp16[t] = (_Float16)fmaxf(acc, 0.0f);
}

// ---------------------------------------------------------------------
// WMMA GEMM:  C[N,128] = A[N,128](f16) @ W[128,128] (+bias) (opt relu)
//   A row-major f16; WT column-major f16 (WT[n*128+k] = W[k][n]).
// B is staged in LDS once per block (32KB of the 320KB WGP LDS), padded to
// stride 136 halves to spread the 16-lane column gather across banks.
// One wave -> 16 rows x 128 cols via 8 accumulators, K-loop of 4.
// A-frag layout (16-bit A 16x32): lane=M%16, h=lane/16,
//   halves 0..7  = K kbase+8h..+7,  halves 8..15 = K kbase+16+8h..+7.
// B-frag layout: lane=N%16, h=lane/16, same K order as A-frag.
// C layout: lane=N%16, VGPR r -> M = r + 8*(lane/16).
__global__ __launch_bounds__(256) void wmmaGemm128K(
    const _Float16* __restrict__ A, const _Float16* __restrict__ WT,
    const float* __restrict__ bias, float* __restrict__ Cf,
    _Float16* __restrict__ Ch, int N, int relu) {
    __shared__ _Float16 sWT[128 * LDS_STRIDE];

    // Cooperative stage of the full 128x128 f16 weight tile into LDS.
    {
        int c  = threadIdx.x >> 1;          // 0..127 : weight column
        int j0 = (threadIdx.x & 1) * 64;    // which half of the K range
        const _Float16* src = WT + (size_t)c * 128 + j0;
        _Float16* dst = &sWT[c * LDS_STRIDE + j0];
#pragma unroll
        for (int j = 0; j < 64; j += 8)
            *(v8h*)(dst + j) = *(const v8h*)(src + j);
    }
    __syncthreads();

    int lane = threadIdx.x & 31;
    int wave = threadIdx.x >> 5;
    int rowBase = blockIdx.x * 128 + wave * 16;
    if (rowBase >= N) return;                 // wave-uniform: EXEC stays full
    int h  = lane >> 4;
    int ln = lane & 15;
    int r  = rowBase + ln;
    if (r > N - 1) r = N - 1;                 // clamp loads; stores are guarded

    v8f acc[8] = {};

#pragma unroll
    for (int kk = 0; kk < 4; ++kk) {
        const _Float16* ap = A + (size_t)r * 128 + kk * 32 + 8 * h;
        v8h a0 = *(const v8h*)ap;             // K = kbase+8h    .. +7
        v8h a1 = *(const v8h*)(ap + 16);      // K = kbase+16+8h .. +7
        v16h a = __builtin_shufflevector(a0, a1,
                 0, 1, 2, 3, 4, 5, 6, 7, 8, 9, 10, 11, 12, 13, 14, 15);
#pragma unroll
        for (int nt = 0; nt < 8; ++nt) {
            const _Float16* bp = &sWT[(nt * 16 + ln) * LDS_STRIDE + kk * 32 + 8 * h];
            v8h b0 = *(const v8h*)bp;         // two 16B ds loads (16B aligned)
            v8h b1 = *(const v8h*)(bp + 16);
            v16h b = __builtin_shufflevector(b0, b1,
                     0, 1, 2, 3, 4, 5, 6, 7, 8, 9, 10, 11, 12, 13, 14, 15);
            acc[nt] = __builtin_amdgcn_wmma_f32_16x16x32_f16(
                false, a, false, b, (short)0, acc[nt], false, false);
        }
    }

#pragma unroll
    for (int nt = 0; nt < 8; ++nt) {
        int col = nt * 16 + ln;
        float bv = bias ? bias[col] : 0.0f;
#pragma unroll
        for (int rr = 0; rr < 8; ++rr) {
            int row = rowBase + rr + 8 * h;
            if (row < N) {
                float v = acc[nt][rr] + bv;
                if (relu) v = fmaxf(v, 0.0f);
                Cf[(size_t)row * 128 + col] = v;
                if (Ch) Ch[(size_t)row * 128 + col] = (_Float16)v;
            }
        }
    }
}

// ---------------------------------------------------------------------
// Attention scores: a_s[i,hd] = <g[i,hd,:], att_src[hd,:]>, same for a_d.
__global__ void scoresK(const float* __restrict__ g,
                        const float* __restrict__ att_s,
                        const float* __restrict__ att_d,
                        float* __restrict__ a_s, float* __restrict__ a_d, int N) {
    int t = blockIdx.x * blockDim.x + threadIdx.x;
    if (t >= N * 4) return;
    int i = t >> 2, hd = t & 3;
    const float* gi = g + (size_t)i * 128 + hd * 32;
    const float* as = att_s + hd * 32;
    const float* ad = att_d + hd * 32;
    float s = 0.0f, d = 0.0f;
#pragma unroll
    for (int c = 0; c < 32; ++c) { s += gi[c] * as[c]; d += gi[c] * ad[c]; }
    a_s[t] = s; a_d[t] = d;
}

// ---------------------------------------------------------------------
__global__ void gatInitK(float* __restrict__ outacc, float* __restrict__ denom,
                         float* __restrict__ mmax, int N) {
    int t = blockIdx.x * blockDim.x + threadIdx.x;
    if (t < N * 128) outacc[t] = 0.0f;
    if (t < N * 4) { denom[t] = 0.0f; mmax[t] = __int_as_float(0xFF800000); }
}

// ---------------------------------------------------------------------
// Edge pass 1: running max of leaky_relu(a_s[src]+a_d[dst]) per (dst, head).
// Virtual edges [E, E+N) are the PyG self-loops.
__global__ void edgeMaxK(const long long* __restrict__ ei,
                         const float* __restrict__ a_s,
                         const float* __restrict__ a_d,
                         float* __restrict__ mmax, int E, int N) {
    int t = blockIdx.x * blockDim.x + threadIdx.x;
    int total = (E + N) * 4;
    if (t >= total) return;
    int hd = t & 3;
    int e  = t >> 2;
    int s, d;
    if (e < E) { s = (int)ei[e]; d = (int)ei[E + e]; }
    else       { s = d = e - E; }
    float v = a_s[s * 4 + hd] + a_d[d * 4 + hd];
    v = (v > 0.0f) ? v : NEG_SLOPE * v;
    atomicMaxF(&mmax[d * 4 + hd], v);
}

// ---------------------------------------------------------------------
// Edge pass 2: p = exp(e - m[dst]); denom += p; out[dst] += p * g[src].
// Prefetch the 128B source feature row (global_prefetch_b8) so the gather
// overlaps the softmax arithmetic.
__global__ void edgeScatterK(const long long* __restrict__ ei,
                             const float* __restrict__ a_s,
                             const float* __restrict__ a_d,
                             const float* __restrict__ mmax,
                             const float* __restrict__ g,
                             float* __restrict__ denom,
                             float* __restrict__ outacc, int E, int N) {
    int t = blockIdx.x * blockDim.x + threadIdx.x;
    int total = (E + N) * 4;
    if (t >= total) return;
    int hd = t & 3;
    int e  = t >> 2;
    int s, d;
    if (e < E) { s = (int)ei[e]; d = (int)ei[E + e]; }
    else       { s = d = e - E; }
    const float* gp = g + (size_t)s * 128 + hd * 32;
    __builtin_prefetch(gp, 0, 0);
    float v = a_s[s * 4 + hd] + a_d[d * 4 + hd];
    v = (v > 0.0f) ? v : NEG_SLOPE * v;
    float p = __expf(v - mmax[d * 4 + hd]);
    unsafeAtomicAdd(&denom[d * 4 + hd], p);
    float* op = outacc + (size_t)d * 128 + hd * 32;
#pragma unroll 8
    for (int c = 0; c < 32; ++c) unsafeAtomicAdd(op + c, gp[c] * p);
}

// ---------------------------------------------------------------------
__global__ void gatFinalizeK(const float* __restrict__ outacc,
                             const float* __restrict__ denom,
                             const float* __restrict__ bias,
                             float* __restrict__ hf, _Float16* __restrict__ hh,
                             int N, int relu) {
    int t = blockIdx.x * blockDim.x + threadIdx.x;
    if (t >= N * 128) return;
    int i = t >> 7, c = t & 127;
    float v = outacc[t] / denom[i * 4 + (c >> 5)] + bias[c];
    if (relu) v = fmaxf(v, 0.0f);
    hf[t] = v;
    hh[t] = (_Float16)v;
}

// ---------------------------------------------------------------------
// Heads tail: logits[N,4] = t[:,0:64]@ed_w2 + ed_b2 ; rep[N] = t[:,64:128]@rp_w2 + rp_b2
// d_out layout: [N*4 logits][N repair]
__global__ void headsK(const float* __restrict__ t,
                       const float* __restrict__ ed_w2, const float* __restrict__ ed_b2,
                       const float* __restrict__ rp_w2, const float* __restrict__ rp_b2,
                       float* __restrict__ out, int N) {
    int i = blockIdx.x * blockDim.x + threadIdx.x;
    if (i >= N) return;
    const float* ti = t + (size_t)i * 128;
    float a0 = ed_b2[0], a1 = ed_b2[1], a2 = ed_b2[2], a3 = ed_b2[3];
#pragma unroll 8
    for (int k = 0; k < 64; ++k) {
        float tv = ti[k];
        a0 += tv * ed_w2[k * 4 + 0];
        a1 += tv * ed_w2[k * 4 + 1];
        a2 += tv * ed_w2[k * 4 + 2];
        a3 += tv * ed_w2[k * 4 + 3];
    }
    out[(size_t)i * 4 + 0] = a0;
    out[(size_t)i * 4 + 1] = a1;
    out[(size_t)i * 4 + 2] = a2;
    out[(size_t)i * 4 + 3] = a3;
    float r = rp_b2[0];
#pragma unroll 8
    for (int k = 0; k < 64; ++k) r += ti[64 + k] * rp_w2[k];
    out[(size_t)N * 4 + i] = r;
}

// =====================================================================
extern "C" void kernel_launch(void* const* d_in, const int* in_sizes, int n_in,
                              void* d_out, int out_size, void* d_ws, size_t ws_size,
                              hipStream_t stream) {
    const float*     x     = (const float*)d_in[0];
    const long long* ei    = (const long long*)d_in[1];
    const float*     ce_w1 = (const float*)d_in[2];
    const float*     ce_b1 = (const float*)d_in[3];
    const float*     ce_w2 = (const float*)d_in[4];
    const float*     ce_b2 = (const float*)d_in[5];
    const float*     g1_w  = (const float*)d_in[6];
    const float*     g1_as = (const float*)d_in[7];
    const float*     g1_ad = (const float*)d_in[8];
    const float*     g1_b  = (const float*)d_in[9];
    const float*     g2_w  = (const float*)d_in[10];
    const float*     g2_as = (const float*)d_in[11];
    const float*     g2_ad = (const float*)d_in[12];
    const float*     g2_b  = (const float*)d_in[13];
    const float*     ed_w1 = (const float*)d_in[14];
    const float*     ed_b1 = (const float*)d_in[15];
    const float*     ed_w2 = (const float*)d_in[16];
    const float*     ed_b2 = (const float*)d_in[17];
    const float*     rp_w1 = (const float*)d_in[18];
    const float*     rp_b1 = (const float*)d_in[19];
    const float*     rp_w2 = (const float*)d_in[20];
    const float*     rp_b2 = (const float*)d_in[21];

    const int N = in_sizes[0] / 8;
    const int E = in_sizes[1] / 2;

    // ------- workspace carve-out (256B aligned slabs) -------
    size_t off = 0;
    char* base = (char*)d_ws;
    auto alloc = [&](size_t bytes) -> void* {
        void* p = base + off;
        off += (bytes + 255) & ~(size_t)255;
        return p;
    };
    float*    hf32   = (float*)alloc((size_t)N * 128 * 4);
    _Float16* hf16   = (_Float16*)alloc((size_t)N * 128 * 2);
    float*    gf32   = (float*)alloc((size_t)N * 128 * 4);
    _Float16* tmp16  = (_Float16*)alloc((size_t)N * 128 * 2);
    float*    outacc = (float*)alloc((size_t)N * 128 * 4);
    float*    a_s    = (float*)alloc((size_t)N * 4 * 4);
    float*    a_d    = (float*)alloc((size_t)N * 4 * 4);
    float*    mmax   = (float*)alloc((size_t)N * 4 * 4);
    float*    denom  = (float*)alloc((size_t)N * 4 * 4);
    _Float16* ceT    = (_Float16*)alloc(128 * 128 * 2);
    _Float16* g1T    = (_Float16*)alloc(128 * 128 * 2);
    _Float16* g2T    = (_Float16*)alloc(128 * 128 * 2);
    _Float16* edrpT  = (_Float16*)alloc(128 * 128 * 2);
    float*    edrpB  = (float*)alloc(128 * 4);
    (void)ws_size; (void)n_in; (void)out_size;

    const int TB = 256;
    dim3 blk(TB);
    auto gridOf = [&](long long total) { return dim3((unsigned)((total + TB - 1) / TB)); };

    // ------- weight prep -------
    transposeConvertK<<<gridOf(128 * 128), blk, 0, stream>>>(ce_w2, ceT, 128, 0);
    transposeConvertK<<<gridOf(128 * 128), blk, 0, stream>>>(g1_w,  g1T, 128, 0);
    transposeConvertK<<<gridOf(128 * 128), blk, 0, stream>>>(g2_w,  g2T, 128, 0);
    transposeConvertK<<<gridOf(64 * 128),  blk, 0, stream>>>(ed_w1, edrpT, 64, 0);
    transposeConvertK<<<gridOf(64 * 128),  blk, 0, stream>>>(rp_w1, edrpT, 64, 64);
    concatBiasK<<<1, 128, 0, stream>>>(ed_b1, rp_b1, edrpB);

    const dim3 gemmGrid((unsigned)((N + 127) / 128));

    // ------- cell encoder -------
    enc1K<<<gridOf((long long)N * 128), blk, 0, stream>>>(x, ce_w1, ce_b1, tmp16, N);
    wmmaGemm128K<<<gemmGrid, blk, 0, stream>>>(tmp16, ceT, ce_b2, hf32, hf16, N, 0);

    // ------- GAT layers -------
    const long long edgeThreads = (long long)(E + N) * 4;
    struct Layer { const _Float16* wT; const float *as, *ad, *b; int relu; };
    Layer layers[2] = { { g1T, g1_as, g1_ad, g1_b, 1 },
                        { g2T, g2_as, g2_ad, g2_b, 0 } };
    for (int L = 0; L < 2; ++L) {
        wmmaGemm128K<<<gemmGrid, blk, 0, stream>>>(hf16, layers[L].wT, nullptr,
                                                   gf32, nullptr, N, 0);
        scoresK<<<gridOf((long long)N * 4), blk, 0, stream>>>(gf32, layers[L].as,
                                                              layers[L].ad, a_s, a_d, N);
        gatInitK<<<gridOf((long long)N * 128), blk, 0, stream>>>(outacc, denom, mmax, N);
        edgeMaxK<<<gridOf(edgeThreads), blk, 0, stream>>>(ei, a_s, a_d, mmax, E, N);
        edgeScatterK<<<gridOf(edgeThreads), blk, 0, stream>>>(ei, a_s, a_d, mmax, gf32,
                                                              denom, outacc, E, N);
        gatFinalizeK<<<gridOf((long long)N * 128), blk, 0, stream>>>(outacc, denom,
                                                                     layers[L].b, hf32,
                                                                     hf16, N, layers[L].relu);
    }

    // ------- output heads: fused [ed_w1 | rp_w1] GEMM + tail -------
    wmmaGemm128K<<<gemmGrid, blk, 0, stream>>>(hf16, edrpT, edrpB, gf32, nullptr, N, 1);
    headsK<<<gridOf(N), blk, 0, stream>>>(gf32, ed_w2, ed_b2, rp_w2, rp_b2,
                                          (float*)d_out, N);
}
